// BFA_57509612094022
// MI455X (gfx1250) — matrix operand order repaired and checked by reading, not compiled, not run on hardware
//
#include <hip/hip_runtime.h>
#include <hip/hip_bf16.h>

// ---------------------------------------------------------------------------
// CDNA5 (gfx1250) Restormer block. Pixel-major bf16 activations [B][S][C] so
// every WMMA A/B fragment and D store is a contiguous 16B/32B access.
// All 1x1 convs: v_wmma_f32_16x16x32_bf16, f32 accumulate, 4 M-tiles/wave.
// Weights pre-converted to padded bf16 in workspace. ws need: ~455 MB.
// ---------------------------------------------------------------------------

typedef __attribute__((ext_vector_type(16))) __bf16 v16bf;
typedef __attribute__((ext_vector_type(8)))  float  v8f;

union V16 { v16bf v; unsigned short u[16]; uint4 q[2]; };

__device__ __forceinline__ unsigned short f2bf(float f) {
    unsigned int x = __float_as_uint(f);
    unsigned int r = (x + 0x7FFFu + ((x >> 16) & 1u)) >> 16;   // RNE
    return (unsigned short)r;
}
__device__ __forceinline__ float bf2f(unsigned short h) {
    return __uint_as_float(((unsigned int)h) << 16);
}
__device__ __forceinline__ float gelu_tanh(float x) {
    const float k0 = 0.7978845608028654f;  // sqrt(2/pi)
    return 0.5f * x * (1.0f + tanhf(k0 * (x + 0.044715f * x * x * x)));
}

#define CB   8
#define CC   192
#define CH   128
#define CWD  128
#define CS   16384
#define CHID 510
#define CIN2 1020

// ---------------------------------------------------------------------------
// Weight convert: f32 [M][Kreal] -> bf16 [Mpad][Kpad], zero-padded.
// ---------------------------------------------------------------------------
__global__ void cvt_w_k(const float* __restrict__ src, unsigned short* __restrict__ dst,
                        int M, int Kreal, int Mpad, int Kpad) {
    int idx = blockIdx.x * 256 + threadIdx.x;
    if (idx >= Mpad * Kpad) return;
    int m = idx / Kpad, k = idx % Kpad;
    dst[idx] = (m < M && k < Kreal) ? f2bf(src[m * Kreal + k]) : (unsigned short)0;
}

// ---------------------------------------------------------------------------
// LayerNorm over channels of f32 NCHW input -> bf16 pixel-major [B][S][192].
// Block = 64-pixel tile staged in LDS (padded rows: no bank conflicts).
// ---------------------------------------------------------------------------
__global__ void ln_t_k(const float* __restrict__ x, const float* __restrict__ w,
                       const float* __restrict__ bi, unsigned short* __restrict__ yT) {
    __shared__ float sm[CC * 65];
    __shared__ float red[2 * 256];
    __shared__ float mu_s[64], inv_s[64];
    int t = threadIdx.x;
    int b  = blockIdx.x / (CS / 64);
    int p0 = (blockIdx.x % (CS / 64)) * 64;
    const float* xb = x + (size_t)b * CC * CS + p0;

    // Phase A: coalesced float4 load of [192][64] tile into LDS (row pad 65)
    #pragma unroll
    for (int i = 0; i < 12; ++i) {
        int idx4 = i * 256 + t;                    // 3072 float4 total
        int c = idx4 >> 4, p4 = (idx4 & 15) << 2;
        float4 v = *(const float4*)(xb + (size_t)c * CS + p4);
        float* d = &sm[c * 65 + p4];
        d[0] = v.x; d[1] = v.y; d[2] = v.z; d[3] = v.w;
    }
    __syncthreads();

    // Phase B: per-pixel mean/var (4 partial sums per pixel)
    int p = t & 63, part = t >> 6;
    float s = 0.f, ss = 0.f;
    for (int c = part * 48; c < part * 48 + 48; ++c) {
        float v = sm[c * 65 + p];
        s += v; ss += v * v;
    }
    red[t] = s; red[256 + t] = ss;
    __syncthreads();
    if (t < 64) {
        float S1 = red[t] + red[64 + t] + red[128 + t] + red[192 + t];
        float S2 = red[256 + t] + red[320 + t] + red[384 + t] + red[448 + t];
        float mu = S1 * (1.f / CC);
        float var = S2 * (1.f / CC) - mu * mu;
        mu_s[t] = mu; inv_s[t] = rsqrtf(var + 1e-5f);
    }
    __syncthreads();

    // Phase C: normalized transpose write, coalesced along c
    unsigned short* yrow = yT + ((size_t)b * CS + p0) * CC;
    if (t < CC) {
        float wc = w[t], bc = bi[t];
        for (int pp = 0; pp < 64; ++pp) {
            float v = (sm[t * 65 + pp] - mu_s[pp]) * inv_s[pp] * wc + bc;
            yrow[(size_t)pp * CC + t] = f2bf(v);
        }
    }
}

// ---------------------------------------------------------------------------
// GEMM: D[b,m,p] = sum_k Wb[m,k] * Xt[b,p,k].  Wb bf16 [Mpad][Kpad] padded,
// Xt bf16 pixel-major row stride ldX.  Wave = 4 M-tiles x 16 N, K step 32.
// MODE 0: bf16 store to YbT (pixel-major, stride ldD)
// MODE 1: f32 NCHW Yf = R + acc      MODE 2: f32 NCHW Yf += acc
// ---------------------------------------------------------------------------
template <int MODE>
__global__ void gemm_wmma_k(const unsigned short* __restrict__ Wb,
                            const unsigned short* __restrict__ Xt,
                            unsigned short* __restrict__ YbT,
                            float* __restrict__ Yf, const float* __restrict__ R,
                            int M, int Kpad, int ldX, int ldD) {
    int b    = blockIdx.z;
    int mg   = blockIdx.y * 64;
    int wid  = threadIdx.x >> 5, lane = threadIdx.x & 31;
    int col  = lane & 15, half = lane >> 4;
    int p    = blockIdx.x * 128 + wid * 16 + col;

    const unsigned short* Xrow = Xt + ((size_t)b * CS + p) * ldX;
    v8f acc[4] = {};
    for (int k0 = 0; k0 < Kpad; k0 += 32) {
        if (k0 + 32 < Kpad) __builtin_prefetch(Xrow + k0 + 32, 0, 1);
        V16 bb;
        const unsigned short* xp = Xrow + k0 + half * 8;    // two 16B chunks (k 8h.., 16+8h..)
        bb.q[0] = *(const uint4*)(xp);
        bb.q[1] = *(const uint4*)(xp + 16);
        #pragma unroll
        for (int mt = 0; mt < 4; ++mt) {
            V16 a;
            const unsigned short* wp =
                Wb + (size_t)(mg + mt * 16 + col) * Kpad + k0 + half * 8;
            a.q[0] = *(const uint4*)(wp);
            a.q[1] = *(const uint4*)(wp + 16);
            acc[mt] = __builtin_amdgcn_wmma_f32_16x16x32_bf16(
                          false, a.v, false, bb.v, (short)0, acc[mt], false, false);
        }
    }
    #pragma unroll
    for (int mt = 0; mt < 4; ++mt) {
        int mbase = mg + mt * 16 + half * 8;
        if (MODE == 0) {
            size_t rowoff = ((size_t)b * CS + p) * ldD;
            if (mbase + 8 <= M) {
                V16 tmp;
                #pragma unroll
                for (int r = 0; r < 8; ++r) tmp.u[r] = f2bf(acc[mt][r]);
                *(uint4*)(YbT + rowoff + mbase) = tmp.q[0];
            } else {
                #pragma unroll
                for (int r = 0; r < 8; ++r)
                    if (mbase + r < M) YbT[rowoff + mbase + r] = f2bf(acc[mt][r]);
            }
        } else {
            #pragma unroll
            for (int r = 0; r < 8; ++r) {
                int m = mbase + r;
                if (m >= M) continue;
                size_t idx = ((size_t)b * M + m) * CS + p;
                if (MODE == 1) Yf[idx] = R[idx] + acc[mt][r];
                else           Yf[idx] += acc[mt][r];
            }
        }
    }
}

// ---------------------------------------------------------------------------
// Depthwise 3x3 kernels: channel-parallel (coalesced), sliding window of 16
// pixels along w (tiles never cross rows: 128/16=8).
// ---------------------------------------------------------------------------
__global__ void dwq_k(const unsigned short* __restrict__ X, const float* __restrict__ W9,
                      unsigned short* __restrict__ Y, float* __restrict__ ss) {
    const int LD = 384;
    int c = blockIdx.y * 128 + threadIdx.x;        // 0..383
    int b = blockIdx.z;
    int p0 = blockIdx.x * 16, h = p0 >> 7, w0 = p0 & 127;
    const unsigned short* Xb = X + (size_t)b * CS * LD;
    float wk[9];
    #pragma unroll
    for (int i = 0; i < 9; ++i) wk[i] = W9[c * 9 + i];
    auto ld = [&](int wx, int ry) -> float {
        int hy = h - 1 + ry;
        if ((unsigned)wx >= (unsigned)CWD || (unsigned)hy >= (unsigned)CH) return 0.f;
        return bf2f(Xb[((size_t)(hy * CWD + wx)) * LD + c]);
    };
    float colb[3][3];
    #pragma unroll
    for (int ry = 0; ry < 3; ++ry) { colb[0][ry] = ld(w0 - 1, ry); colb[1][ry] = ld(w0, ry); }
    float ssq = 0.f;
    #pragma unroll
    for (int i = 0; i < 16; ++i) {
        #pragma unroll
        for (int ry = 0; ry < 3; ++ry) colb[2][ry] = ld(w0 + i + 1, ry);
        float acc = 0.f;
        #pragma unroll
        for (int ry = 0; ry < 3; ++ry)
            acc += wk[ry * 3] * colb[0][ry] + wk[ry * 3 + 1] * colb[1][ry] + wk[ry * 3 + 2] * colb[2][ry];
        Y[((size_t)b * CS + p0 + i) * LD + c] = f2bf(acc);
        ssq += acc * acc;
        #pragma unroll
        for (int ry = 0; ry < 3; ++ry) { colb[0][ry] = colb[1][ry]; colb[1][ry] = colb[2][ry]; }
    }
    atomicAdd(&ss[b * 384 + c], ssq);
}

// v depthwise -> head-major padded layout [B][S][256] (64 slots/head, 48 valid)
__global__ void dwv_k(const unsigned short* __restrict__ X, const float* __restrict__ W9,
                      unsigned short* __restrict__ Y) {
    int slot = blockIdx.y * 128 + threadIdx.x;     // 0..255
    int b = blockIdx.z;
    int p0 = blockIdx.x * 16, h = p0 >> 7, w0 = p0 & 127;
    int hd = slot >> 6, cc = slot & 63;
    if (cc >= 48) {
        for (int i = 0; i < 16; ++i) Y[((size_t)b * CS + p0 + i) * 256 + slot] = 0;
        return;
    }
    int c = hd * 48 + cc;
    const unsigned short* Xb = X + (size_t)b * CS * CC;
    float wk[9];
    #pragma unroll
    for (int i = 0; i < 9; ++i) wk[i] = W9[c * 9 + i];
    auto ld = [&](int wx, int ry) -> float {
        int hy = h - 1 + ry;
        if ((unsigned)wx >= (unsigned)CWD || (unsigned)hy >= (unsigned)CH) return 0.f;
        return bf2f(Xb[((size_t)(hy * CWD + wx)) * CC + c]);
    };
    float colb[3][3];
    #pragma unroll
    for (int ry = 0; ry < 3; ++ry) { colb[0][ry] = ld(w0 - 1, ry); colb[1][ry] = ld(w0, ry); }
    #pragma unroll
    for (int i = 0; i < 16; ++i) {
        #pragma unroll
        for (int ry = 0; ry < 3; ++ry) colb[2][ry] = ld(w0 + i + 1, ry);
        float acc = 0.f;
        #pragma unroll
        for (int ry = 0; ry < 3; ++ry)
            acc += wk[ry * 3] * colb[0][ry] + wk[ry * 3 + 1] * colb[1][ry] + wk[ry * 3 + 2] * colb[2][ry];
        Y[((size_t)b * CS + p0 + i) * 256 + slot] = f2bf(acc);
        #pragma unroll
        for (int ry = 0; ry < 3; ++ry) { colb[0][ry] = colb[1][ry]; colb[1][ry] = colb[2][ry]; }
    }
}

// FFN depthwise + gelu-gate: pinT [B][S][1024] -> gatedT [B][S][512] (c>=510 zero)
__global__ void ffn_gate_k(const unsigned short* __restrict__ X, const float* __restrict__ W9,
                           unsigned short* __restrict__ G) {
    int c = blockIdx.y * 128 + threadIdx.x;        // 0..511
    int b = blockIdx.z;
    int p0 = blockIdx.x * 16, h = p0 >> 7, w0 = p0 & 127;
    if (c >= CHID) {
        for (int i = 0; i < 16; ++i) G[((size_t)b * CS + p0 + i) * 512 + c] = 0;
        return;
    }
    const unsigned short* Xb = X + (size_t)b * CS * 1024;
    float wa[9], wg[9];
    #pragma unroll
    for (int i = 0; i < 9; ++i) { wa[i] = W9[c * 9 + i]; wg[i] = W9[(c + CHID) * 9 + i]; }
    auto ld = [&](int wx, int ry, int ch) -> float {
        int hy = h - 1 + ry;
        if ((unsigned)wx >= (unsigned)CWD || (unsigned)hy >= (unsigned)CH) return 0.f;
        return bf2f(Xb[((size_t)(hy * CWD + wx)) * 1024 + ch]);
    };
    float ca[3][3], cg[3][3];
    #pragma unroll
    for (int ry = 0; ry < 3; ++ry) {
        ca[0][ry] = ld(w0 - 1, ry, c); ca[1][ry] = ld(w0, ry, c);
        cg[0][ry] = ld(w0 - 1, ry, c + CHID); cg[1][ry] = ld(w0, ry, c + CHID);
    }
    #pragma unroll
    for (int i = 0; i < 16; ++i) {
        #pragma unroll
        for (int ry = 0; ry < 3; ++ry) { ca[2][ry] = ld(w0 + i + 1, ry, c); cg[2][ry] = ld(w0 + i + 1, ry, c + CHID); }
        float aA = 0.f, aG = 0.f;
        #pragma unroll
        for (int ry = 0; ry < 3; ++ry) {
            aA += wa[ry * 3] * ca[0][ry] + wa[ry * 3 + 1] * ca[1][ry] + wa[ry * 3 + 2] * ca[2][ry];
            aG += wg[ry * 3] * cg[0][ry] + wg[ry * 3 + 1] * cg[1][ry] + wg[ry * 3 + 2] * cg[2][ry];
        }
        G[((size_t)b * CS + p0 + i) * 512 + c] = f2bf(gelu_tanh(aA) * aG);
        #pragma unroll
        for (int ry = 0; ry < 3; ++ry) {
            ca[0][ry] = ca[1][ry]; ca[1][ry] = ca[2][ry];
            cg[0][ry] = cg[1][ry]; cg[1][ry] = cg[2][ry];
        }
    }
}

__global__ void zero_f32_k(float* p, int n) {
    int i = blockIdx.x * 256 + threadIdx.x;
    if (i < n) p[i] = 0.f;
}
__global__ void make_scale_k(const float* __restrict__ ss, float* __restrict__ sc, int n) {
    int i = blockIdx.x * 256 + threadIdx.x;
    if (i < n) sc[i] = 1.0f / fmaxf(sqrtf(ss[i]), 1e-12f);
}

// ---------------------------------------------------------------------------
// Gram + softmax per (b,head): G = q_dw @ k_dw^T over K=16384 spatial, WMMA.
// qkdwT pixel-major [B][S][384] (strided-but-coalesced fragment gathers).
// A = softmax(temp*qs_i*ks_j*G) -> bf16 [b,h,48,64] zero-padded.
// ---------------------------------------------------------------------------
__global__ void attn_gram_k(const unsigned short* __restrict__ QK,
                            const float* __restrict__ scale,
                            const float* __restrict__ temp,
                            unsigned short* __restrict__ A) {
    int b = blockIdx.x >> 2, hd = blockIdx.x & 3;
    int wid = threadIdx.x >> 5, lane = threadIdx.x & 31;
    int col = lane & 15, half = lane >> 4;
    __shared__ float G[48][48];
    const unsigned short* qb = QK + (size_t)b * CS * 384;

    for (int t = wid; t < 9; t += 8) {
        int mi = t / 3, ni = t % 3;
        int qc = hd * 48 + mi * 16 + col;
        int kc = 192 + hd * 48 + ni * 16 + col;
        v8f acc = {};
        for (int k0 = 0; k0 < CS; k0 += 32) {
            V16 a, bb;
            int base = k0 + half * 8;
            #pragma unroll
            for (int e = 0; e < 8; ++e) {
                a.u[e]     = qb[(size_t)(base + e) * 384 + qc];
                a.u[8 + e] = qb[(size_t)(base + 16 + e) * 384 + qc];
            }
            int kb = k0 + half * 8;
            #pragma unroll
            for (int e = 0; e < 16; ++e) bb.u[e] = qb[(size_t)(kb + e + (e & 8)) * 384 + kc];
            acc = __builtin_amdgcn_wmma_f32_16x16x32_bf16(
                      false, a.v, false, bb.v, (short)0, acc, false, false);
        }
        #pragma unroll
        for (int r = 0; r < 8; ++r)
            G[mi * 16 + half * 8 + r][ni * 16 + col] = acc[r];
    }
    __syncthreads();

    int i = threadIdx.x;
    if (i < 48) {
        float qs = scale[b * 384 + hd * 48 + i];
        float tp = temp[hd];
        float row[48], mx = -1e30f;
        for (int j = 0; j < 48; ++j) {
            float v = tp * qs * scale[b * 384 + 192 + hd * 48 + j] * G[i][j];
            row[j] = v; mx = fmaxf(mx, v);
        }
        float sum = 0.f;
        for (int j = 0; j < 48; ++j) { float e = __expf(row[j] - mx); row[j] = e; sum += e; }
        float inv = 1.0f / sum;
        unsigned short* Ar = A + ((size_t)(b * 4 + hd) * 48 + i) * 64;
        for (int j = 0; j < 48; ++j) Ar[j] = f2bf(row[j] * inv);
        for (int j = 48; j < 64; ++j) Ar[j] = 0;
    }
}

// ---------------------------------------------------------------------------
// out = A @ v_dw : M=48, K=64 (padded), N=16384 per (b,head).
// V padded head-major [B][S][256] -> unconditional contiguous B fragments.
// ---------------------------------------------------------------------------
__global__ void attn_av_k(const unsigned short* __restrict__ A,
                          const unsigned short* __restrict__ Vp,
                          unsigned short* __restrict__ OT) {
    int bz = blockIdx.z, b = bz >> 2, hd = bz & 3;
    int m0 = blockIdx.y * 16;
    int wid = threadIdx.x >> 5, lane = threadIdx.x & 31;
    int col = lane & 15, half = lane >> 4;
    int p = blockIdx.x * 128 + wid * 16 + col;

    const unsigned short* ar = A + ((size_t)(b * 4 + hd) * 48 + m0 + col) * 64;
    const unsigned short* vrow = Vp + ((size_t)b * CS + p) * 256 + hd * 64;

    v8f acc = {};
    #pragma unroll
    for (int k0 = 0; k0 < 64; k0 += 32) {
        V16 a, bb;
        const unsigned short* ap = ar + k0 + half * 8;
        a.q[0] = *(const uint4*)(ap);
        a.q[1] = *(const uint4*)(ap + 16);
        const unsigned short* vp2 = vrow + k0 + half * 8;
        bb.q[0] = *(const uint4*)(vp2);
        bb.q[1] = *(const uint4*)(vp2 + 16);
        acc = __builtin_amdgcn_wmma_f32_16x16x32_bf16(
                  false, a.v, false, bb.v, (short)0, acc, false, false);
    }
    V16 tmp;
    #pragma unroll
    for (int r = 0; r < 8; ++r) tmp.u[r] = f2bf(acc[r]);
    *(uint4*)(OT + ((size_t)b * CS + p) * CC + hd * 48 + m0 + half * 8) = tmp.q[0];
}

// ---------------------------------------------------------------------------
extern "C" void kernel_launch(void* const* d_in, const int* in_sizes, int n_in,
                              void* d_out, int out_size, void* d_ws, size_t ws_size,
                              hipStream_t stream) {
    (void)in_sizes; (void)n_in; (void)out_size; (void)ws_size;
    const float* x       = (const float*)d_in[0];
    const float* ln1_w   = (const float*)d_in[1];
    const float* ln1_b   = (const float*)d_in[2];
    const float* qk_w    = (const float*)d_in[3];
    const float* qkdw_w  = (const float*)d_in[4];
    const float* v_w     = (const float*)d_in[5];
    const float* vdw_w   = (const float*)d_in[6];
    const float* proj_w  = (const float*)d_in[7];
    const float* temp    = (const float*)d_in[8];
    const float* ln2_w   = (const float*)d_in[9];
    const float* ln2_b   = (const float*)d_in[10];
    const float* pin_w   = (const float*)d_in[11];
    const float* ffndw_w = (const float*)d_in[12];
    const float* pout_w  = (const float*)d_in[13];
    float* out = (float*)d_out;

    // ---- workspace carving ----
    char* ws = (char*)d_ws;
    const size_t szT0  = (size_t)CB * CS * CC  * 2;   //  50,331,648
    const size_t szBIG = (size_t)CB * CS * (1024 + 512) * 2; // 402,653,184
    unsigned short* t0T  = (unsigned short*)(ws);
    char* big = ws + szT0;
    // phase 1 overlay
    unsigned short* qkT   = (unsigned short*)(big);
    unsigned short* qkdwT = (unsigned short*)(big + (size_t)CB * CS * 384 * 2);
    unsigned short* vT    = (unsigned short*)(big + (size_t)CB * CS * 768 * 2);
    unsigned short* vdwpT = (unsigned short*)(big + (size_t)CB * CS * 960 * 2);
    unsigned short* aoutT = (unsigned short*)(big + (size_t)CB * CS * 1216 * 2);
    // phase 2 overlay
    unsigned short* pinT   = (unsigned short*)(big);
    unsigned short* gatedT = (unsigned short*)(big + (size_t)CB * CS * 1024 * 2);
    // weights (bf16, padded) + small buffers
    char* wend = ws + szT0 + szBIG;
    unsigned short* qkWb   = (unsigned short*)(wend);                        // 384*192
    unsigned short* vWb    = qkWb  + 384 * 192;                              // 192*192
    unsigned short* projWb = vWb   + 192 * 192;                              // 192*192
    unsigned short* pinWb  = projWb + 192 * 192;                             // 1024*192
    unsigned short* poutWb = pinWb + 1024 * 192;                             // 192*512
    unsigned short* attnA  = poutWb + 192 * 512;                             // 32*48*64
    float* ssbuf = (float*)(attnA + 32 * 48 * 64);
    float* scbuf = ssbuf + CB * 384;

    const dim3 blk(256), blk128(128);

    // 0. weight conversion (tiny)
    cvt_w_k<<<dim3((384 * 192 + 255) / 256), blk, 0, stream>>>(qk_w, qkWb, 384, 192, 384, 192);
    cvt_w_k<<<dim3((192 * 192 + 255) / 256), blk, 0, stream>>>(v_w, vWb, 192, 192, 192, 192);
    cvt_w_k<<<dim3((192 * 192 + 255) / 256), blk, 0, stream>>>(proj_w, projWb, 192, 192, 192, 192);
    cvt_w_k<<<dim3((1024 * 192 + 255) / 256), blk, 0, stream>>>(pin_w, pinWb, 1020, 192, 1024, 192);
    cvt_w_k<<<dim3((192 * 512 + 255) / 256), blk, 0, stream>>>(pout_w, poutWb, 192, 510, 192, 512);

    // 1. LN1 -> t0T
    ln_t_k<<<dim3(CB * CS / 64), blk, 0, stream>>>(x, ln1_w, ln1_b, t0T);

    // 2. qk = W_qk @ t0
    gemm_wmma_k<0><<<dim3(CS / 128, 6, CB), blk, 0, stream>>>(
        qkWb, t0T, qkT, nullptr, nullptr, 384, 192, 192, 384);

    // 3. depthwise qk + fused sum-of-squares -> scales
    zero_f32_k<<<dim3((CB * 384 + 255) / 256), blk, 0, stream>>>(ssbuf, CB * 384);
    dwq_k<<<dim3(CS / 16, 3, CB), blk128, 0, stream>>>(qkT, qkdw_w, qkdwT, ssbuf);
    make_scale_k<<<dim3((CB * 384 + 255) / 256), blk, 0, stream>>>(ssbuf, scbuf, CB * 384);

    // 4-5. v path (padded head-major output)
    gemm_wmma_k<0><<<dim3(CS / 128, 3, CB), blk, 0, stream>>>(
        vWb, t0T, vT, nullptr, nullptr, 192, 192, 192, 192);
    dwv_k<<<dim3(CS / 16, 2, CB), blk128, 0, stream>>>(vT, vdw_w, vdwpT);

    // 6. gram + softmax
    attn_gram_k<<<dim3(CB * 4), blk, 0, stream>>>(qkdwT, scbuf, temp, attnA);

    // 7. out = A @ v_dw
    attn_av_k<<<dim3(CS / 128, 3, CB * 4), blk, 0, stream>>>(attnA, vdwpT, aoutT);

    // 8. d_out = x + proj @ aout
    gemm_wmma_k<1><<<dim3(CS / 128, 3, CB), blk, 0, stream>>>(
        projWb, aoutT, nullptr, out, x, 192, 192, 192, 0);

    // 9. LN2 -> t0T
    ln_t_k<<<dim3(CB * CS / 64), blk, 0, stream>>>(out, ln2_w, ln2_b, t0T);

    // 10. pin
    gemm_wmma_k<0><<<dim3(CS / 128, 16, CB), blk, 0, stream>>>(
        pinWb, t0T, pinT, nullptr, nullptr, 1020, 192, 192, 1024);

    // 11. depthwise + gelu-gate
    ffn_gate_k<<<dim3(CS / 16, 4, CB), blk128, 0, stream>>>(pinT, ffndw_w, gatedT);

    // 12. d_out += pout @ gated
    gemm_wmma_k<2><<<dim3(CS / 128, 3, CB), blk, 0, stream>>>(
        poutWb, gatedT, nullptr, out, nullptr, 192, 512, 512, 0);
}
